// Attention2_83751862272066
// MI455X (gfx1250) — matrix operand order repaired and checked by reading, not compiled, not run on hardware
//
#include <hip/hip_runtime.h>
#include <hip/hip_bf16.h>

// ---------------------------------------------------------------------------
// Problem constants (from reference): B=64, S=257, D=768, H=12, HD=64
// Branch token counts: 1x1 -> 257, 3x3 VALID -> 197, 5x5 VALID -> 145
// ---------------------------------------------------------------------------
#define NB   64
#define NS   257
#define ND   768
#define NH   12
#define NHD  64
#define TOK_TOTAL 599   // 257 + 197 + 145 per batch

typedef __attribute__((ext_vector_type(16))) _Float16 v16h;
typedef __attribute__((ext_vector_type(8)))  float    v8f;

union F16x16 { v16h v; _Float16 h[16]; };

// Guarded gfx1250 async-LDS path (verified present on this toolchain).
#if defined(__HIP_DEVICE_COMPILE__) && defined(__gfx1250__)
#if __has_builtin(__builtin_amdgcn_global_load_async_to_lds_b128) && \
    __has_builtin(__builtin_amdgcn_s_wait_asynccnt)
#define ATH_ASYNC_LDS 1
#endif
#endif

#if defined(ATH_ASYNC_LDS)
#define ATH_WAIT_ASYNC() __builtin_amdgcn_s_wait_asynccnt(0)
#else
#define ATH_WAIT_ASYNC()
#endif

// ---------------------------------------------------------------------------
// f32 -> f16 convert (grid-stride)
// ---------------------------------------------------------------------------
__global__ void cvt_f32_f16(const float* __restrict__ src,
                            _Float16* __restrict__ dst, size_t n) {
  size_t i = (size_t)blockIdx.x * blockDim.x + threadIdx.x;
  size_t stride = (size_t)gridDim.x * blockDim.x;
  for (; i < n; i += stride) dst[i] = (_Float16)src[i];
}

// ---------------------------------------------------------------------------
// Build the 599 tokens per batch: branch0 = 1x1 depthwise conv on the 16x16
// grid (+cls), branch1 = 3x3 VALID (14x14, +cls), branch2 = 5x5 VALID
// (12x12, +cls).  Output packed f16 [B, 599, 768].
// ---------------------------------------------------------------------------
__global__ __launch_bounds__(256) void build_tokens(
    const float* __restrict__ hs, const float* __restrict__ c1,
    const float* __restrict__ c2, const float* __restrict__ c3,
    _Float16* __restrict__ xb) {
  const int t = blockIdx.x;  // 0..598
  const int b = blockIdx.y;  // 0..63
  for (int d = threadIdx.x; d < ND; d += 256) {
    float val;
    if (t < 257) {                       // branch 0: 1x1 conv
      if (t == 0) {
        val = hs[((size_t)b * NS) * ND + d];
      } else {
        val = hs[((size_t)b * NS + t) * ND + d] * c1[d];
      }
    } else if (t < 454) {                // branch 1: 3x3 VALID -> 14x14
      const int lt = t - 257;
      if (lt == 0) {
        val = hs[((size_t)b * NS) * ND + d];
      } else {
        const int p = lt - 1, oy = p / 14, ox = p % 14;
        float acc = 0.f;
        for (int ky = 0; ky < 3; ++ky)
          for (int kx = 0; kx < 3; ++kx)
            acc += hs[((size_t)b * NS + 1 + (oy + ky) * 16 + (ox + kx)) * ND + d]
                 * c2[d * 9 + ky * 3 + kx];
        val = acc;
      }
    } else {                             // branch 2: 5x5 VALID -> 12x12
      const int lt = t - 454;
      if (lt == 0) {
        val = hs[((size_t)b * NS) * ND + d];
      } else {
        const int p = lt - 1, oy = p / 12, ox = p % 12;
        float acc = 0.f;
        for (int ky = 0; ky < 5; ++ky)
          for (int kx = 0; kx < 5; ++kx)
            acc += hs[((size_t)b * NS + 1 + (oy + ky) * 16 + (ox + kx)) * ND + d]
                 * c3[d * 25 + ky * 5 + kx];
        val = acc;
      }
    }
    xb[((size_t)b * TOK_TOTAL + t) * ND + d] = (_Float16)val;
  }
}

// ---------------------------------------------------------------------------
// WMMA GEMM:  Y[M,768] = X[M,768] @ W^T + bias
//   Block = 8 waves = one 128(M) x 64(N) block tile; blockIdx.y = n-group.
//   The 64x32 weight tile per K-step is staged once per block into LDS via
//   async-LDS loads (double buffered), shared by all 8 waves; A fragments
//   stream from global.  96 WMMAs per wave.
//   X rows are segmented per batch: row m -> (b = m/Ltok, t = m%Ltok),
//   actual X row = (b*batchTokens + tokOfs + t).
//   mode 0: f32 row-major out (final projection)
//   mode 1: f16 out at [ (b*H+h)*outTokStride + outTokOfs + t ][hd]   (Q, K)
//   mode 2: f16 out at [ (b*H+h)*64 + hd ][outTokOfs + t]          (V^T)
// ---------------------------------------------------------------------------
__global__ __launch_bounds__(256) void gemm_wmma(
    const _Float16* __restrict__ X, const _Float16* __restrict__ Wt,
    const float* __restrict__ bias,
    int M, int Ltok, int batchTokens, int tokOfs,
    int mode, _Float16* __restrict__ outH, float* __restrict__ outF,
    int outTokStride, int outTokOfs) {
  __shared__ _Float16 ldsW[2][64 * 32];  // [n(64)][k(32)] double-buffered

  const int tid  = threadIdx.x;
  const int lane = tid & 31;
  const int wave = tid >> 5;
  const int ngrp = blockIdx.y;                 // 0..11 (64 cols each)
  const int mtile = blockIdx.x * 8 + wave;     // 16-row tile index
  const bool valid = mtile < (M >> 4);

  const int lm = lane & 15;  // A row / B,C column within tile
  const int lh = lane >> 4;  // half selector

  // Per-lane A row pointer (A fragment: row = lane&15); clamp when invalid.
  const int mg = min(mtile * 16 + lm, M - 1);
  const int bA = mg / Ltok;
  const int tA = mg % Ltok;
  const _Float16* arow =
      X + ((size_t)(bA * batchTokens + tokOfs + tA)) * ND;

  // Stage one 64x32 weight tile into LDS (4KB: 256 threads x 16B).
  auto stageW = [&](int kc, int bufi) {
    const int n  = tid >> 2;         // 0..63
    const int ko = (tid & 3) << 3;   // 0,8,16,24
    const _Float16* src = Wt + (size_t)(ngrp * 64 + n) * ND + kc + ko;
    _Float16* dst = &ldsW[bufi][n * 32 + ko];
#if defined(ATH_ASYNC_LDS)
    typedef int __attribute__((ext_vector_type(4))) int4v;
    typedef __attribute__((address_space(1))) int4v gint4v;
    typedef __attribute__((address_space(3))) int4v lint4v;
    __builtin_amdgcn_global_load_async_to_lds_b128(
        (gint4v*)(unsigned long long)src,
        (lint4v*)(unsigned)(unsigned long long)dst, 0, 0);
#else
#pragma unroll
    for (int e = 0; e < 8; ++e) dst[e] = src[e];
#endif
  };

  stageW(0, 0);
  ATH_WAIT_ASYNC();
  __syncthreads();

  v8f acc[4] = {};
  int cur = 0;
  for (int kc = 0; kc < ND; kc += 32) {
    if (kc + 32 < ND) stageW(kc + 32, cur ^ 1);  // async prefetch next tile
    __builtin_prefetch(arow + kc + 64, 0, 1);    // -> global_prefetch_b8
    F16x16 a;
#pragma unroll
    for (int e = 0; e < 16; ++e)
      a.h[e] = arow[kc + lh * 16 + e];
#pragma unroll
    for (int nt = 0; nt < 4; ++nt) {
      // B fragment from LDS: col = lane&15, k = lh*16+e (ds_load_b128 x2)
      F16x16 bf;
#pragma unroll
      for (int e = 0; e < 16; ++e)
        bf.h[e] = ldsW[cur][(nt * 16 + lm) * 32 + lh * 16 + e];
      acc[nt] = __builtin_amdgcn_wmma_f32_16x16x32_f16(
          false, a.v, false, bf.v, (short)0, acc[nt], false, false);
    }
    ATH_WAIT_ASYNC();   // next tile landed (per-wave asynccnt)
    __syncthreads();    // everyone done reading cur, next buffer complete
    cur ^= 1;
  }

  if (!valid) return;

  // C layout: row = r + 8*lh, col = lane&15.
#pragma unroll
  for (int nt = 0; nt < 4; ++nt) {
    const int col = ngrp * 64 + nt * 16 + lm;
    const float bv = bias[col];
#pragma unroll
    for (int r = 0; r < 8; ++r) {
      const int mrow = mtile * 16 + r + 8 * lh;
      const float val = acc[nt][r] + bv;
      if (mode == 0) {
        outF[(size_t)mrow * ND + col] = val;
      } else {
        const int bI = mrow / Ltok, tI = mrow % Ltok;
        const int hI = col >> 6, dI = col & 63;
        if (mode == 1)
          outH[(((size_t)(bI * NH + hI)) * outTokStride + outTokOfs + tI) * NHD + dI] =
              (_Float16)val;
        else
          outH[(((size_t)(bI * NH + hI)) * NHD + dI) * outTokStride + outTokOfs + tI] =
              (_Float16)val;
      }
    }
  }
}

// ---------------------------------------------------------------------------
// Flash-style attention, 1 wave per (q-tile, b*h).  Loops the 3 branches
// internally (independent softmaxes), accumulates ctx = (sum_br O_br)/3 into
// f16 [B*S, 768] so the final GEMM can consume it directly.
// Q: [B,H,257,64]  K: [B,H,599,64] (branch segments)  VT: [B,H,64,599]
// ---------------------------------------------------------------------------
__global__ __launch_bounds__(32) void attn_wmma(
    const _Float16* __restrict__ Q, const _Float16* __restrict__ K,
    const _Float16* __restrict__ VT, _Float16* __restrict__ ctx) {
  __shared__ _Float16 ldsK[32 * 64];   // staged K chunk
  __shared__ _Float16 ldsP[16 * 32];   // P tile relayout (C-frag -> A-frag)

  const int lane = threadIdx.x;
  const int lm = lane & 15, lh = lane >> 4;
  const int qt = blockIdx.x;           // 0..16
  const int bh = blockIdx.y;           // 0..767
  const int b = bh / NH, h = bh % NH;

  const _Float16* qbase = Q + (size_t)bh * NS * NHD;
  const _Float16* kbase = K + (size_t)bh * TOK_TOTAL * NHD;
  const _Float16* vtb   = VT + (size_t)bh * NHD * TOK_TOTAL;

  // Q A-fragments (16 rows x 64 hd = two 16x32 fragments), loaded once.
  F16x16 aq[2];
  {
    const int qrow = min(qt * 16 + lm, NS - 1);
    const _Float16* qr = qbase + (size_t)qrow * NHD;
#pragma unroll
    for (int hc = 0; hc < 2; ++hc)
#pragma unroll
      for (int e = 0; e < 16; ++e)
        aq[hc].h[e] = qr[hc * 32 + lh * 16 + e];
  }

  const float scale = 0.125f;  // 1/sqrt(64)
  const int Ls[3] = {257, 197, 145};
  const int Os[3] = {0, 257, 454};

  v8f octx[4] = {};

  for (int br = 0; br < 3; ++br) {
    const int L = Ls[br], ofs = Os[br];
    v8f o[4] = {};
    float mrun[8], lrun[8];
#pragma unroll
    for (int r = 0; r < 8; ++r) { mrun[r] = -1e30f; lrun[r] = 0.f; }

    const int nchunks = (L + 31) >> 5;
    for (int jc = 0; jc < nchunks; ++jc) {
      __syncthreads();  // protect LDS reuse from previous iteration
      // ---- stage K chunk [32 kv x 64 hd] (4 KB, contiguous) into LDS ----
      const _Float16* ksrc = kbase + ((size_t)(ofs + jc * 32)) * NHD;
#if defined(ATH_ASYNC_LDS)
      {
        typedef int __attribute__((ext_vector_type(4))) int4v;
        typedef __attribute__((address_space(1))) int4v gint4v;
        typedef __attribute__((address_space(3))) int4v lint4v;
        gint4v* gsrc = (gint4v*)(unsigned long long)ksrc;
        lint4v* ldst = (lint4v*)(unsigned)(unsigned long long)&ldsK[0];
#pragma unroll
        for (int it = 0; it < 8; ++it) {
          const int seg = it * 32 + lane;     // 16-byte segments (4KB total)
          __builtin_amdgcn_global_load_async_to_lds_b128(gsrc + seg,
                                                         ldst + seg, 0, 0);
        }
        __builtin_amdgcn_s_wait_asynccnt(0);
      }
#else
      for (int it = 0; it < 8; ++it) {
        const int seg = it * 32 + lane;
#pragma unroll
        for (int e = 0; e < 8; ++e) ldsK[seg * 8 + e] = ksrc[seg * 8 + e];
      }
#endif
      __syncthreads();

      // ---- scores: S[16 x 32] = Q(16x64) . K_chunk^T, two C fragments ----
      v8f s0 = {}, s1 = {};
#pragma unroll
      for (int hc = 0; hc < 2; ++hc) {
        F16x16 kb0, kb1;  // B fragments: col = kv row (lane&15), k = hd
#pragma unroll
        for (int e = 0; e < 16; ++e) {
          kb0.h[e] = ldsK[lm * NHD + hc * 32 + lh * 16 + e];
          kb1.h[e] = ldsK[(16 + lm) * NHD + hc * 32 + lh * 16 + e];
        }
        s0 = __builtin_amdgcn_wmma_f32_16x16x32_f16(
            false, aq[hc].v, false, kb0.v, (short)0, s0, false, false);
        s1 = __builtin_amdgcn_wmma_f32_16x16x32_f16(
            false, aq[hc].v, false, kb1.v, (short)0, s1, false, false);
      }

      // scale + mask invalid kv columns
      const int c0 = jc * 32 + lm, c1 = c0 + 16;
#pragma unroll
      for (int r = 0; r < 8; ++r) {
        s0[r] = (c0 < L) ? s0[r] * scale : -1e30f;
        s1[r] = (c1 < L) ? s1[r] * scale : -1e30f;
      }

      // ---- online softmax (row groups live in 16-lane halves) ----
#pragma unroll
      for (int r = 0; r < 8; ++r) {
        float cm = fmaxf(s0[r], s1[r]);
        for (int off = 8; off; off >>= 1)
          cm = fmaxf(cm, __shfl_xor(cm, off, 16));
        const float nm = fmaxf(mrun[r], cm);
        const float corr = __expf(mrun[r] - nm);
        const float p0 = __expf(s0[r] - nm);
        const float p1 = __expf(s1[r] - nm);
        float rs = p0 + p1;
        for (int off = 8; off; off >>= 1) rs += __shfl_xor(rs, off, 16);
        lrun[r] = lrun[r] * corr + rs;
        mrun[r] = nm;
        o[0][r] *= corr; o[1][r] *= corr; o[2][r] *= corr; o[3][r] *= corr;
        const int row = r + 8 * lh;
        ldsP[row * 32 + lm]      = (_Float16)p0;
        ldsP[row * 32 + 16 + lm] = (_Float16)p1;
      }
      __syncthreads();

      // ---- O += P(16x32) . V_chunk(32x64) ----
      F16x16 pa;  // A fragment of P
#pragma unroll
      for (int e = 0; e < 16; ++e)
        pa.h[e] = ldsP[lm * 32 + lh * 16 + e];
#pragma unroll
      for (int nt = 0; nt < 4; ++nt) {
        const int hd = nt * 16 + lm;
        F16x16 vb;  // B fragment: col = hd, k = kv row (contiguous via V^T)
#pragma unroll
        for (int e = 0; e < 16; ++e) {
          int t = jc * 32 + lh * 16 + e;
          t = (t < L) ? t : (L - 1);
          vb.h[e] = vtb[(size_t)hd * TOK_TOTAL + ofs + t];
        }
        o[nt] = __builtin_amdgcn_wmma_f32_16x16x32_f16(
            false, pa.v, false, vb.v, (short)0, o[nt], false, false);
      }
    }

    // normalize this branch and accumulate
#pragma unroll
    for (int nt = 0; nt < 4; ++nt)
#pragma unroll
      for (int r = 0; r < 8; ++r) octx[nt][r] += o[nt][r] / lrun[r];
  }

  // store ctx/3 as f16 [B*S, 768]; C layout: row = r + 8*lh, col = lane&15
#pragma unroll
  for (int nt = 0; nt < 4; ++nt) {
    const int col = h * NHD + nt * 16 + lm;
#pragma unroll
    for (int r = 0; r < 8; ++r) {
      const int s = qt * 16 + r + 8 * lh;
      if (s < NS)
        ctx[((size_t)(b * NS + s)) * ND + col] =
            (_Float16)(octx[nt][r] * (1.0f / 3.0f));
    }
  }
}

// ---------------------------------------------------------------------------
// Host-side orchestration
// ---------------------------------------------------------------------------
extern "C" void kernel_launch(void* const* d_in, const int* in_sizes, int n_in,
                              void* d_out, int out_size, void* d_ws,
                              size_t ws_size, hipStream_t stream) {
  const float* hs = (const float*)d_in[0];
  const float* Wq = (const float*)d_in[1];
  const float* bq = (const float*)d_in[2];
  const float* Wk = (const float*)d_in[3];
  const float* bk = (const float*)d_in[4];
  const float* Wv = (const float*)d_in[5];
  const float* bv = (const float*)d_in[6];
  const float* Wo = (const float*)d_in[7];
  const float* bo = (const float*)d_in[8];
  const float* c1 = (const float*)d_in[9];
  const float* c2 = (const float*)d_in[10];
  const float* c3 = (const float*)d_in[11];

  // workspace layout (all sizes 256B-aligned)
  const size_t szHS  = (size_t)NB * NS * ND * 2;              // hs f16
  const size_t szW   = (size_t)ND * ND * 2;                   // one weight f16
  const size_t szQ   = (size_t)NB * NS * ND * 2;              // q f16 [B,H,S,HD]
  const size_t szXB  = (size_t)NB * TOK_TOTAL * ND * 2;       // tokens f16
  const size_t szK   = (size_t)NB * NH * TOK_TOTAL * NHD * 2; // k f16
  const size_t szVT  = szK;                                   // v^T f16
  const size_t szCTX = (size_t)NB * NS * ND * 2;              // ctx f16

  char* ws = (char*)d_ws;
  _Float16* hsF  = (_Float16*)(ws);
  _Float16* wqF  = (_Float16*)(ws + szHS);
  _Float16* wkF  = (_Float16*)(ws + szHS + szW);
  _Float16* wvF  = (_Float16*)(ws + szHS + 2 * szW);
  _Float16* woF  = (_Float16*)(ws + szHS + 3 * szW);
  _Float16* qF   = (_Float16*)(ws + szHS + 4 * szW);
  _Float16* xbF  = (_Float16*)(ws + szHS + 4 * szW + szQ);
  _Float16* kF   = (_Float16*)(ws + szHS + 4 * szW + szQ + szXB);
  _Float16* vtF  = (_Float16*)(ws + szHS + 4 * szW + szQ + szXB + szK);
  _Float16* ctxF = (_Float16*)(ws + szHS + 4 * szW + szQ + szXB + szK + szVT);
  if (ws_size < szHS + 4 * szW + szQ + szXB + szK + szVT + szCTX) return;

  // 1) converts
  {
    const size_t nHS = (size_t)NB * NS * ND;
    cvt_f32_f16<<<4096, 256, 0, stream>>>(hs, hsF, nHS);
    const size_t nW = (size_t)ND * ND;
    cvt_f32_f16<<<1024, 256, 0, stream>>>(Wq, wqF, nW);
    cvt_f32_f16<<<1024, 256, 0, stream>>>(Wk, wkF, nW);
    cvt_f32_f16<<<1024, 256, 0, stream>>>(Wv, wvF, nW);
    cvt_f32_f16<<<1024, 256, 0, stream>>>(Wo, woF, nW);
  }

  // 2) conv branches -> packed tokens
  build_tokens<<<dim3(TOK_TOTAL, NB), 256, 0, stream>>>(hs, c1, c2, c3, xbF);

  const int Ls[3] = {257, 197, 145};
  const int Os[3] = {0, 257, 454};

  // 3) Q projection -> [B,H,S,HD] f16
  {
    const int M = NB * NS;
    const int mblocks = ((M / 16) + 7) / 8;
    gemm_wmma<<<dim3(mblocks, ND / 64), 256, 0, stream>>>(
        hsF, wqF, bq, M, NS, NS, 0, /*mode=*/1, qF, nullptr, NS, 0);
  }

  // 4) K / V projections per branch
  for (int br = 0; br < 3; ++br) {
    const int L = Ls[br], ofs = Os[br];
    const int M = NB * L;
    const int mblocks = ((M / 16) + 7) / 8;
    gemm_wmma<<<dim3(mblocks, ND / 64), 256, 0, stream>>>(
        xbF, wkF, bk, M, L, TOK_TOTAL, ofs, /*mode=*/1, kF, nullptr,
        TOK_TOTAL, ofs);
    gemm_wmma<<<dim3(mblocks, ND / 64), 256, 0, stream>>>(
        xbF, wvF, bv, M, L, TOK_TOTAL, ofs, /*mode=*/2, vtF, nullptr,
        TOK_TOTAL, ofs);
  }

  // 5) attention (all 3 branches fused, writes ctx/3 as f16 [B*S, 768])
  attn_wmma<<<dim3((NS + 15) / 16, NB * NH), 32, 0, stream>>>(qF, kF, vtF,
                                                              ctxF);

  // 6) output projection -> f32 d_out
  {
    const int M = NB * NS;
    const int mblocks = ((M / 16) + 7) / 8;
    gemm_wmma<<<dim3(mblocks, ND / 64), 256, 0, stream>>>(
        ctxF, woF, bo, M, NS, NS, 0, /*mode=*/0, nullptr, (float*)d_out, NS,
        0);
  }
}
